// ScaledDotProductAttention_64252710748224
// MI455X (gfx1250) — matrix operand order repaired and checked by reading, not compiled, not run on hardware
//
#include <hip/hip_runtime.h>

// ---------------------------------------------------------------------------
// SDPA (out, att) for gfx1250 / MI455X, bf16 WMMA (v_wmma_f32_16x16x32_bf16)
// Two-pass flash-style softmax so `att` is written normalized exactly once.
// Pass-1 softmax stats are lane-local; one (m,l) butterfly merge after the
// K sweep. fp32->bf16 uses v_cvt_pk_bf16_f32 when the builtin is available.
// ---------------------------------------------------------------------------

typedef __attribute__((ext_vector_type(16))) __bf16 v16bf;
typedef __attribute__((ext_vector_type(8)))  float  v8f;

#define L_SEQ 8192
#define DIM   128
#define BM    64            // query rows per workgroup (4 waves x 16 rows)
#define BN    128           // keys per K/V tile
#define NW    4             // waves per workgroup
#define NTHR  (NW * 32)
#define LDK   136           // bf16 elems per LDS row (128 + 8 pad, 16B-aligned rows)

static __device__ __forceinline__ unsigned short bf16_bits_sw(float f) {
  unsigned u = __builtin_bit_cast(unsigned, f);
  unsigned r = u + 0x7FFFu + ((u >> 16) & 1u);     // round-to-nearest-even
  return (unsigned short)(r >> 16);
}

// Pack two fp32 into two bf16 (RNE). Prefers v_cvt_pk_bf16_f32.
static __device__ __forceinline__ unsigned pack2_bf16(float a, float b) {
#if __has_builtin(__builtin_amdgcn_cvt_pk_bf16_f32)
  typedef __attribute__((ext_vector_type(2))) __bf16 v2bf;
  v2bf p = __builtin_amdgcn_cvt_pk_bf16_f32(a, b);
  return __builtin_bit_cast(unsigned, p);
#else
  return (unsigned)bf16_bits_sw(a) | ((unsigned)bf16_bits_sw(b) << 16);
#endif
}
static __device__ __forceinline__ unsigned short bf16_low(float a) {
  return (unsigned short)(pack2_bf16(a, a) & 0xFFFFu);
}

// A-matrix (16x32 bf16) per-lane slice from an LDS row:
// lanes 0-15: regs 0-7 = K base+0..15 (lanes 16-31: base+16)
static __device__ __forceinline__ v16bf lds_a_operand(const unsigned short* row, int kbase) {
  union { uint4 q[2]; v16bf v; } u;
  u.q[0] = *(const uint4*)(row + kbase);
  u.q[1] = *(const uint4*)(row + kbase + 8);
  return u.v;
}
// B-matrix (32x16 bf16) per-lane slice: lane holds column, 16 contiguous K values
static __device__ __forceinline__ v16bf lds_b_operand(const unsigned short* row, int kbase) {
  union { uint4 q[2]; v16bf v; } u;
  u.q[0] = *(const uint4*)(row + kbase);
  u.q[1] = *(const uint4*)(row + kbase + 8);
  return u.v;
}

// stage 128x128 fp32 row-major tile -> LDS bf16 [n][LDK]
static __device__ __forceinline__ void stage_rowmajor(unsigned short* dst, const float* src, int tid) {
  for (int i = tid; i < BN * (DIM / 4); i += NTHR) {
    int n  = i >> 5;
    int k4 = (i & 31) << 2;
    float4 f = *(const float4*)(src + (size_t)n * DIM + k4);
    uint2 pk;
    pk.x = pack2_bf16(f.x, f.y);
    pk.y = pack2_bf16(f.z, f.w);
    *(uint2*)(dst + n * LDK + k4) = pk;
  }
}
// stage K row-major and V transposed in one sweep (pass 2)
static __device__ __forceinline__ void stage_kv(unsigned short* dk, const float* srck,
                                                unsigned short* dv, const float* srcv, int tid) {
  for (int i = tid; i < BN * (DIM / 4); i += NTHR) {
    int n  = i >> 5;          // row (key index)
    int k4 = (i & 31) << 2;   // column base
    float4 f = *(const float4*)(srck + (size_t)n * DIM + k4);
    uint2 pk;
    pk.x = pack2_bf16(f.x, f.y);
    pk.y = pack2_bf16(f.z, f.w);
    *(uint2*)(dk + n * LDK + k4) = pk;

    float4 g = *(const float4*)(srcv + (size_t)n * DIM + k4);
    dv[(k4 + 0) * LDK + n] = bf16_low(g.x);
    dv[(k4 + 1) * LDK + n] = bf16_low(g.y);
    dv[(k4 + 2) * LDK + n] = bf16_low(g.z);
    dv[(k4 + 3) * LDK + n] = bf16_low(g.w);
  }
}

__global__ __launch_bounds__(NTHR) void
sdpa_kernel(const float* __restrict__ Q, const float* __restrict__ Km,
            const float* __restrict__ Vm, float* __restrict__ out,
            float* __restrict__ att) {
  extern __shared__ unsigned short smem[];
  unsigned short* Kb = smem;                     // BN * LDK  (K tile, [key][k])
  unsigned short* Vb = Kb + BN * LDK;            // BN * LDK  (V tile transposed, [col][key])
  unsigned short* Pb = Vb + BN * LDK;            // NW * 16 * LDK (per-wave P bounce)

  const int tid   = threadIdx.x;
  const int wave  = tid >> 5;
  const int lane  = tid & 31;
  const int l16   = lane & 15;
  const int lhalf = lane >> 4;
  const int qbase = blockIdx.x * BM + wave * 16;

  const float cc = 0.1275237559041565f;          // (1/sqrt(128)) * log2(e)

  // ---- resident Q operand in A-layout (16 rows x 128 K, 4 k-steps)
  v16bf qa[4];
  {
    const float* qrow = Q + (size_t)(qbase + l16) * DIM;
    #pragma unroll
    for (int c = 0; c < 4; ++c) {
      union { unsigned u[8]; v16bf v; } uq;
      #pragma unroll
      for (int j = 0; j < 8; ++j) {
        int k = c * 32 + lhalf * 16 + 2 * j;
        uq.u[j] = pack2_bf16(qrow[k], qrow[k + 1]);
      }
      qa[c] = uq.v;
    }
  }

  // Lane-local online softmax stats: mrow[r]/lrow[r] cover this lane's columns
  // of row (r + 8*lhalf). Cross-lane merge happens once, after the K sweep.
  float mrow[8], lrow[8];
  #pragma unroll
  for (int r = 0; r < 8; ++r) { mrow[r] = -1e30f; lrow[r] = 0.0f; }

  // ================= pass 1: lane-local online softmax statistics ==========
  for (int j0 = 0; j0 < L_SEQ; j0 += BN) {
    __syncthreads();
    stage_rowmajor(Kb, Km + (size_t)j0 * DIM, tid);
    if (j0 + BN < L_SEQ)
      __builtin_prefetch(Km + (size_t)(j0 + BN) * DIM + tid * 16, 0, 1);
    __syncthreads();

    v8f sv[8] = {};
    #pragma unroll
    for (int t = 0; t < BN / 16; ++t) {
      const unsigned short* brow = Kb + (t * 16 + l16) * LDK;
      #pragma unroll
      for (int c = 0; c < 4; ++c) {
        v16bf b = lds_b_operand(brow, c * 32 + lhalf * 16);
        sv[t] = __builtin_amdgcn_wmma_f32_16x16x32_bf16(false, qa[c], false, b,
                                                        (short)0, sv[t], false, false);
      }
    }
    #pragma unroll
    for (int r = 0; r < 8; ++r) {
      float zt[BN / 16];
      float tm = -1e30f;
      #pragma unroll
      for (int t = 0; t < BN / 16; ++t) {
        zt[t] = sv[t][r] * cc;
        tm = fmaxf(tm, zt[t]);
      }
      float mn = fmaxf(mrow[r], tm);
      float sum = 0.0f;
      #pragma unroll
      for (int t = 0; t < BN / 16; ++t)
        sum += __builtin_amdgcn_exp2f(zt[t] - mn);
      lrow[r] = lrow[r] * __builtin_amdgcn_exp2f(mrow[r] - mn) + sum;
      mrow[r] = mn;
    }
  }

  // ---- single cross-lane (m,l) merge: 16-lane butterfly per row
  #pragma unroll
  for (int r = 0; r < 8; ++r) {
    float m = mrow[r], l = lrow[r];
    #pragma unroll
    for (int x = 1; x < 16; x <<= 1) {
      float mo = __shfl_xor(m, x, 16);
      float lo = __shfl_xor(l, x, 16);
      float mn = fmaxf(m, mo);
      l = l * __builtin_amdgcn_exp2f(m - mn) + lo * __builtin_amdgcn_exp2f(mo - mn);
      m = mn;
    }
    mrow[r] = m;
    lrow[r] = 1.0f / l;      // store reciprocal; lrow now holds inv_l
  }

  // ================= pass 2: write att, accumulate out = P @ V =============
  v8f acc[8] = {};
  unsigned short* Pw = Pb + wave * 16 * LDK;

  for (int j0 = 0; j0 < L_SEQ; j0 += BN) {
    __syncthreads();
    stage_kv(Kb, Km + (size_t)j0 * DIM, Vb, Vm + (size_t)j0 * DIM, tid);
    if (j0 + BN < L_SEQ) {
      __builtin_prefetch(Km + (size_t)(j0 + BN) * DIM + tid * 16, 0, 1);
      __builtin_prefetch(Vm + (size_t)(j0 + BN) * DIM + tid * 16, 0, 1);
    }
    __syncthreads();

    #pragma unroll
    for (int t = 0; t < BN / 16; ++t) {
      v8f s = {};
      const unsigned short* brow = Kb + (t * 16 + l16) * LDK;
      #pragma unroll
      for (int c = 0; c < 4; ++c) {
        v16bf b = lds_b_operand(brow, c * 32 + lhalf * 16);
        s = __builtin_amdgcn_wmma_f32_16x16x32_bf16(false, qa[c], false, b,
                                                    (short)0, s, false, false);
      }
      #pragma unroll
      for (int r = 0; r < 8; ++r) {
        float p = __builtin_amdgcn_exp2f(s[r] * cc - mrow[r]) * lrow[r];
        int row_l = r + 8 * lhalf;                 // C-layout: lanes16-31 hold M+8
        att[(size_t)(qbase + row_l) * L_SEQ + (j0 + t * 16 + l16)] = p;
        Pw[row_l * LDK + t * 16 + l16] = bf16_low(p);
      }
    }
    __syncthreads();   // order P stores vs. operand gathers (and align waves)

    v16bf pa[4];
    const unsigned short* prow = Pw + l16 * LDK;
    #pragma unroll
    for (int c = 0; c < 4; ++c)
      pa[c] = lds_a_operand(prow, c * 32 + lhalf * 16);

    #pragma unroll
    for (int t = 0; t < DIM / 16; ++t) {
      const unsigned short* vrow = Vb + (t * 16 + l16) * LDK;
      #pragma unroll
      for (int c = 0; c < 4; ++c) {
        v16bf b = lds_b_operand(vrow, c * 32 + lhalf * 16);
        acc[t] = __builtin_amdgcn_wmma_f32_16x16x32_bf16(false, pa[c], false, b,
                                                         (short)0, acc[t], false, false);
      }
    }
  }

  #pragma unroll
  for (int t = 0; t < DIM / 16; ++t) {
    #pragma unroll
    for (int r = 0; r < 8; ++r)
      out[(size_t)(qbase + r + 8 * lhalf) * DIM + t * 16 + l16] = acc[t][r];
  }
}

extern "C" void kernel_launch(void* const* d_in, const int* in_sizes, int n_in,
                              void* d_out, int out_size, void* d_ws, size_t ws_size,
                              hipStream_t stream) {
  (void)in_sizes; (void)n_in; (void)out_size; (void)d_ws; (void)ws_size;
  const float* Q = (const float*)d_in[0];
  const float* K = (const float*)d_in[1];
  const float* V = (const float*)d_in[2];
  float* out = (float*)d_out;
  float* att = out + (size_t)L_SEQ * DIM;   // tuple (out, att) concatenated flat

  size_t shmem = (size_t)(2 * BN + NW * 16) * LDK * sizeof(unsigned short); // 87,040 B
  dim3 grid(L_SEQ / BM);                    // 128 workgroups
  sdpa_kernel<<<grid, NTHR, shmem, stream>>>(Q, K, V, out, att);
}
